// TransformerBlock_42554535969089
// MI455X (gfx1250) — compile-verified, matrix operand order checked by reading, not run on hardware
//
#include <hip/hip_runtime.h>
#include <math.h>

// ---------------- problem dims ----------------
constexpr int Bq   = 2;
constexpr int Lctx = 1024;
constexpr int Dm   = 1024;
constexpr int Hn   = 16;
constexpr int HDim = 64;     // Dm / Hn
constexpr int LATd = 16;
constexpr int NE   = 8;
constexpr int NS   = 2;
constexpr int HIDd = 4096;
constexpr int NTOK = Bq * Lctx;          // 2048
constexpr int ND   = NTOK * Dm;          // 2,097,152
constexpr int NQC  = NTOK * Hn * LATd;   // 524,288  (== B*H*L*LAT)

// ---------------- WMMA types ----------------
typedef __attribute__((ext_vector_type(16))) __bf16 v16bf;
typedef __attribute__((ext_vector_type(8)))  float  v8f;

struct __align__(16) U4 { unsigned x, y, z, w; };
struct __align__(8)  U2 { unsigned x, y; };
union FragU { v16bf v; U4 q[2]; };

__device__ inline unsigned short f2bf(float f) {
  unsigned u = __float_as_uint(f);
  unsigned r = u + 0x7FFFu + ((u >> 16) & 1u);   // round-to-nearest-even
  return (unsigned short)(r >> 16);
}

// pack two fp32 -> two bf16 in one u32 (manual RNE: beats the builtin's expansion on gfx1250)
__device__ inline unsigned pack_bf16x2(float a, float b) {
  return (unsigned)f2bf(a) | ((unsigned)f2bf(b) << 16);
}

enum { EP_STORE = 0, EP_RESID = 1, EP_GELU = 2, EP_ACCUM = 3 };

__device__ inline float gelu_exact(float x) {
  return 0.5f * x * (1.f + erff(x * 0.70710678118654752f));
}

template <int EP, bool BOUND>
__device__ inline void store_tile(float* __restrict__ out, const float* __restrict__ bias,
                                  const float* __restrict__ resid, v8f c,
                                  int R, int C, int rows, int M, int lr, int hh, float scale) {
  // C/D layout: col = C + (lane&15); row = R + vgpr + 8*(lane>>4)
  const int col = C + lr;
  const float bv = bias[col];
#pragma unroll
  for (int v = 0; v < 8; ++v) {
    int row = R + hh * 8 + v;
    if (BOUND && row >= rows) continue;
    float val = c[v] + bv;
    if (EP == EP_GELU) val = gelu_exact(val);
    size_t o = (size_t)row * M + col;
    if (EP == EP_RESID) val += resid[o];
    if (EP == EP_ACCUM) out[o] += scale * val;
    else                out[o] = val;
  }
}

// C[rows, M] = A[rows, Kd] @ W[Kd, M] + bias, epilogue per EP.
// fp32 in/out, bf16 WMMA compute (v_wmma_f32_16x16x32_bf16), fp32 accum.
// Block: 256 thr (8 wave32), tile 128 rows x 128 cols, K-step 32.
// Each wave owns 32 rows x 64 cols = 8 WMMA tiles per k-step.
// Software-pipelined: next k-tile global loads overlap current WMMA block.
template <int EP, bool BOUND>
__global__ void __launch_bounds__(256)
gemm_wmma(const float* __restrict__ A, const float* __restrict__ W,
          const float* __restrict__ bias, const float* __restrict__ resid,
          float* __restrict__ out, int nrows, const int* __restrict__ cntp,
          int Kd, int M, float scale) {
  __shared__ unsigned short As[128][40];   // [row][k]  pitch 40 keeps 16B align
  __shared__ unsigned short Bs[128][40];   // [col][k]  (B staged transposed)

  const int tid  = threadIdx.x;
  const int rows = BOUND ? cntp[0] : nrows;  // device-side row count (MoE routing)
  const int rb   = blockIdx.y * 128;
  if (BOUND && rb >= rows) return;           // whole block exits -> no barrier hazard
  const int cb   = blockIdx.x * 128;

  const int lane = tid & 31;
  const int wave = tid >> 5;
  const int wr   = wave & 3;     // 4 row-groups of 32
  const int wc   = wave >> 2;    // 2 col-groups of 64
  const int lr   = lane & 15;
  const int hh   = lane >> 4;

  v8f acc[2][4];
#pragma unroll
  for (int i = 0; i < 2; ++i)
#pragma unroll
    for (int j = 0; j < 4; ++j) acc[i][j] = v8f{0.f,0.f,0.f,0.f,0.f,0.f,0.f,0.f};

  const int la_r = tid >> 3;         // 0..31 (A loader row within 32-row slab)
  const int la_c = (tid & 7) << 2;   // 0,4,...,28 (A loader k offset)

  float4 ga[4];   // staged A tile (128x32): 4 x float4 per thread
  float4 gb[4];   // staged B tile (32x128): 4 x float4 per thread

  auto load_global = [&](int k0) {
#pragma unroll
    for (int p = 0; p < 4; ++p) {
      int g = rb + p * 32 + la_r;
      if (!BOUND || g < rows)
        ga[p] = *(const float4*)(A + (size_t)g * Kd + (k0 + la_c));
      else { ga[p].x = 0.f; ga[p].y = 0.f; ga[p].z = 0.f; ga[p].w = 0.f; }
    }
#pragma unroll
    for (int p = 0; p < 4; ++p) {
      int e  = (p * 256 + tid) << 2;   // 0..4095 over 32x128 tile
      int kk = e >> 7;                 // 0..31
      int mm = e & 127;                // multiple of 4
      gb[p] = *(const float4*)(W + (size_t)(k0 + kk) * M + (cb + mm));
    }
  };

  load_global(0);
  for (int k0 = 0; k0 < Kd; k0 += 32) {
    __syncthreads();   // previous compute done reading LDS
    // commit staged tile to LDS (fp32 -> bf16; A path uses 8B LDS stores)
#pragma unroll
    for (int p = 0; p < 4; ++p) {
      int row = p * 32 + la_r;
      U2 w2; w2.x = pack_bf16x2(ga[p].x, ga[p].y);
      w2.y = pack_bf16x2(ga[p].z, ga[p].w);
      *(U2*)&As[row][la_c] = w2;       // byte offset row*80 + la_c*2, 8B aligned
    }
#pragma unroll
    for (int p = 0; p < 4; ++p) {
      int e  = (p * 256 + tid) << 2;
      int kk = e >> 7;
      int mm = e & 127;
      Bs[mm + 0][kk] = f2bf(gb[p].x);
      Bs[mm + 1][kk] = f2bf(gb[p].y);
      Bs[mm + 2][kk] = f2bf(gb[p].z);
      Bs[mm + 3][kk] = f2bf(gb[p].w);
    }
    __syncthreads();

    if (k0 + 32 < Kd) load_global(k0 + 32);   // overlap next tile with WMMAs
    if (k0 + 64 < Kd)                          // warm L2 for the tile after next
      __builtin_prefetch(W + (size_t)(k0 + 64 + (tid >> 4)) * M + cb + ((tid & 15) << 3), 0, 1);

    // A frags (16-bit 16x32 layout): lane row = lane&15; K = hh*8+j and 16+hh*8+j
    FragU a0, a1;
    const int ra = wr * 32 + lr;
    a0.q[0] = *(const U4*)&As[ra][hh * 8];
    a0.q[1] = *(const U4*)&As[ra][16 + hh * 8];
    a1.q[0] = *(const U4*)&As[ra + 16][hh * 8];
    a1.q[1] = *(const U4*)&As[ra + 16][16 + hh * 8];
    const int ca = wc * 64 + lr;
#pragma unroll
    for (int tj = 0; tj < 4; ++tj) {
      FragU b;
      b.q[0] = *(const U4*)&Bs[ca + tj * 16][hh * 8];
      b.q[1] = *(const U4*)&Bs[ca + tj * 16][16 + hh * 8];
      acc[0][tj] = __builtin_amdgcn_wmma_f32_16x16x32_bf16(false, a0.v, false, b.v, (short)0, acc[0][tj], false, false);
      acc[1][tj] = __builtin_amdgcn_wmma_f32_16x16x32_bf16(false, a1.v, false, b.v, (short)0, acc[1][tj], false, false);
    }
  }

#pragma unroll
  for (int ti = 0; ti < 2; ++ti)
#pragma unroll
    for (int tj = 0; tj < 4; ++tj)
      store_tile<EP, BOUND>(out, bias, resid, acc[ti][tj],
                            rb + wr * 32 + ti * 16, cb + wc * 64 + tj * 16,
                            rows, M, lr, hh, scale);
}

// ---------------- LayerNorm: one token per 256-thread block ----------------
__global__ void __launch_bounds__(256)
ln_kernel(const float* __restrict__ x, const float* __restrict__ g,
          const float* __restrict__ b, float* __restrict__ out) {
  __shared__ float red[256];
  const int n = blockIdx.x;
  const float* xr = x + (size_t)n * Dm;
  float s = 0.f;
  for (int i = threadIdx.x; i < Dm; i += 256) s += xr[i];
  red[threadIdx.x] = s; __syncthreads();
  for (int o = 128; o > 0; o >>= 1) { if (threadIdx.x < o) red[threadIdx.x] += red[threadIdx.x + o]; __syncthreads(); }
  const float mean = red[0] * (1.f / Dm);
  __syncthreads();
  float v = 0.f;
  for (int i = threadIdx.x; i < Dm; i += 256) { float d = xr[i] - mean; v += d * d; }
  red[threadIdx.x] = v; __syncthreads();
  for (int o = 128; o > 0; o >>= 1) { if (threadIdx.x < o) red[threadIdx.x] += red[threadIdx.x + o]; __syncthreads(); }
  const float rstd = rsqrtf(red[0] * (1.f / Dm) + 1e-5f);
  for (int i = threadIdx.x; i < Dm; i += 256)
    out[(size_t)n * Dm + i] = (xr[i] - mean) * rstd * g[i] + b[i];
}

// ---------------- RoPE in place on [NTOK, Dm] ----------------
__global__ void rope_kernel(float* __restrict__ t, const float* __restrict__ cosb,
                            const float* __restrict__ sinb) {
  int idx = blockIdx.x * 256 + threadIdx.x;         // NTOK*Hn*32
  if (idx >= NTOK * Hn * (HDim / 2)) return;
  const int i = idx & 31;
  const int h = (idx >> 5) & (Hn - 1);
  const int n = idx >> 9;
  const int l = n & (Lctx - 1);
  const float c = cosb[l * (HDim / 2) + i];
  const float s = sinb[l * (HDim / 2) + i];
  float* p = t + (size_t)n * Dm + h * HDim + 2 * i;
  const float te = p[0], to = p[1];
  p[0] = te * c - to * s;
  p[1] = te * s + to * c;
}

// ---------------- latent projection: dst[bh,l,lat] = src[n, h*64: ] @ W[64,16] + b ----------------
__global__ void latproj_kernel(const float* __restrict__ src, const float* __restrict__ W,
                               const float* __restrict__ bias, float* __restrict__ dst) {
  int idx = blockIdx.x * 256 + threadIdx.x;         // NQC
  if (idx >= NQC) return;
  const int lat = idx & 15;
  const int l   = (idx >> 4) & (Lctx - 1);
  const int bh  = idx >> 14;
  const int b   = bh >> 4, h = bh & 15;
  const float* row = src + (size_t)(b * Lctx + l) * Dm + h * HDim;
  float a = bias[lat];
#pragma unroll 8
  for (int hd = 0; hd < HDim; ++hd) a += row[hd] * W[hd * LATd + lat];
  dst[idx] = a;
}

// ---------------- causal latent attention + Wdc epilogue ----------------
__global__ void __launch_bounds__(128)
attn_kernel(const float* __restrict__ qc, const float* __restrict__ kc,
            const float* __restrict__ vc, const float* __restrict__ Wdc,
            const float* __restrict__ bdc, float* __restrict__ attnh) {
  __shared__ float kct[128][16];
  __shared__ float vct[128][16];
  __shared__ float wdc_s[16][64];
  const int bh = blockIdx.y;
  const int b  = bh / Hn, h = bh % Hn;
  const int l  = blockIdx.x * 128 + threadIdx.x;
  for (int i = threadIdx.x; i < 16 * 64; i += 128) wdc_s[i >> 6][i & 63] = Wdc[i];

  float qr[16];
  const float* qrow = qc + ((size_t)bh * Lctx + l) * LATd;
#pragma unroll
  for (int j = 0; j < 16; ++j) qr[j] = qrow[j];

  float m = -3.4e38f, ssum = 0.f, acc[16];
#pragma unroll
  for (int j = 0; j < 16; ++j) acc[j] = 0.f;

  const int ktiles = blockIdx.x + 1;
  for (int kt = 0; kt < ktiles; ++kt) {
    __syncthreads();
    const float* kp = kc + ((size_t)bh * Lctx + kt * 128) * LATd;
    const float* vp = vc + ((size_t)bh * Lctx + kt * 128) * LATd;
    for (int i = threadIdx.x; i < 128 * 16; i += 128) {
      kct[i >> 4][i & 15] = kp[i];
      vct[i >> 4][i & 15] = vp[i];
    }
    __syncthreads();
    int kmax = l - kt * 128 + 1;
    if (kmax > 128) kmax = 128;
    for (int kk = 0; kk < kmax; ++kk) {
      float s = 0.f;
#pragma unroll
      for (int j = 0; j < 16; ++j) s += qr[j] * kct[kk][j];
      s *= 0.25f;                                   // 1/sqrt(LAT)
      const float nm = fmaxf(m, s);
      const float corr = __expf(m - nm);
      const float w = __expf(s - nm);
      ssum = ssum * corr + w;
#pragma unroll
      for (int j = 0; j < 16; ++j) acc[j] = acc[j] * corr + w * vct[kk][j];
      m = nm;
    }
  }
  const float inv = 1.f / ssum;
#pragma unroll
  for (int j = 0; j < 16; ++j) acc[j] *= inv;
  const size_t n = (size_t)b * Lctx + l;
  for (int hd = 0; hd < 64; ++hd) {
    float o = bdc[hd];
#pragma unroll
    for (int j = 0; j < 16; ++j) o += acc[j] * wdc_s[j][hd];
    attnh[n * Dm + h * HDim + hd] = o;
  }
}

// ---------------- router gates ----------------
__global__ void gate_kernel(const float* __restrict__ h2, const float* __restrict__ Wr,
                            const float* __restrict__ br, float* __restrict__ gates) {
  int idx = blockIdx.x * 256 + threadIdx.x;
  if (idx >= NTOK * NE) return;
  const int n = idx / NE, e = idx % NE;
  const float* hr = h2 + (size_t)n * Dm;
  float g = br[e];
  for (int d = 0; d < Dm; ++d) g += hr[d] * Wr[d * NE + e];
  gates[idx] = g;
}

// ---------------- top-2 + expert compaction lists ----------------
__global__ void topk_kernel(const float* __restrict__ gates, int* __restrict__ cnt,
                            int* __restrict__ eidx, float* __restrict__ ew) {
  int n = blockIdx.x * 256 + threadIdx.x;
  if (n >= NTOK) return;
  float g[NE];
#pragma unroll
  for (int e = 0; e < NE; ++e) g[e] = gates[n * NE + e];
  int i0 = 0;
#pragma unroll
  for (int e = 1; e < NE; ++e) if (g[e] > g[i0]) i0 = e;
  int i1 = (i0 == 0) ? 1 : 0;
#pragma unroll
  for (int e = 0; e < NE; ++e) if (e != i0 && g[e] > g[i1]) i1 = e;
  const float z  = __expf(g[i1] - g[i0]);
  const float p0 = 1.f / (1.f + z);
  const float p1 = z / (1.f + z);
  int s0 = atomicAdd(&cnt[i0], 1); eidx[i0 * NTOK + s0] = n; ew[i0 * NTOK + s0] = p0;
  int s1 = atomicAdd(&cnt[i1], 1); eidx[i1 * NTOK + s1] = n; ew[i1 * NTOK + s1] = p1;
}

// ---------------- MoE gather / scatter (per expert, race-free) ----------------
__global__ void gather_kernel(const float* __restrict__ h2, const int* __restrict__ cntp,
                              const int* __restrict__ eidx, float* __restrict__ h2g) {
  const int slot = blockIdx.y;
  if (slot >= cntp[0]) return;
  const int n = eidx[slot];
  const int d = blockIdx.x * 256 + threadIdx.x;
  h2g[(size_t)slot * Dm + d] = h2[(size_t)n * Dm + d];
}

__global__ void scatter_kernel(const float* __restrict__ oeg, const int* __restrict__ cntp,
                               const int* __restrict__ eidx, const float* __restrict__ ew,
                               float* __restrict__ acc) {
  const int slot = blockIdx.y;
  if (slot >= cntp[0]) return;
  const int n = eidx[slot];
  const float w = ew[slot];
  const int d = blockIdx.x * 256 + threadIdx.x;
  acc[(size_t)n * Dm + d] += w * oeg[(size_t)slot * Dm + d];
}

// ---------------- misc elementwise ----------------
__global__ void zero_kernel(float* __restrict__ acc, int* __restrict__ cnt) {
  const int i = blockIdx.x * 256 + threadIdx.x;
  if (i < ND) acc[i] = 0.f;
  if (blockIdx.x == 0 && threadIdx.x < 64) cnt[threadIdx.x] = 0;
}

__global__ void final_add_kernel(const float* __restrict__ x2, const float* __restrict__ acc,
                                 float* __restrict__ out) {
  const int i = blockIdx.x * 256 + threadIdx.x;
  if (i < ND) out[i] = x2[i] + acc[i];
}

// =====================================================================
extern "C" void kernel_launch(void* const* d_in, const int* in_sizes, int n_in,
                              void* d_out, int out_size, void* d_ws, size_t ws_size,
                              hipStream_t stream) {
  const float* x    = (const float*)d_in[0];
  const float* cosb = (const float*)d_in[1];
  const float* sinb = (const float*)d_in[2];
  const float* ln1g = (const float*)d_in[3];
  const float* ln1b = (const float*)d_in[4];
  const float* Wq   = (const float*)d_in[5];
  const float* bq   = (const float*)d_in[6];
  const float* Wk   = (const float*)d_in[7];
  const float* bk   = (const float*)d_in[8];
  const float* Wv   = (const float*)d_in[9];
  const float* bv   = (const float*)d_in[10];
  const float* Wqc  = (const float*)d_in[11];
  const float* bqc  = (const float*)d_in[12];
  const float* Wkc  = (const float*)d_in[13];
  const float* bkc  = (const float*)d_in[14];
  const float* Wvc  = (const float*)d_in[15];
  const float* bvc  = (const float*)d_in[16];
  const float* Wdc  = (const float*)d_in[17];
  const float* bdc  = (const float*)d_in[18];
  const float* Wo   = (const float*)d_in[19];
  const float* bo   = (const float*)d_in[20];
  const float* ln2g = (const float*)d_in[21];
  const float* ln2b = (const float*)d_in[22];
  const float* Wr   = (const float*)d_in[23];
  const float* br   = (const float*)d_in[24];
  const float* We1  = (const float*)d_in[25];
  const float* be1  = (const float*)d_in[26];
  const float* We2  = (const float*)d_in[27];
  const float* be2  = (const float*)d_in[28];
  const float* Ws1  = (const float*)d_in[29];
  const float* bs1  = (const float*)d_in[30];
  const float* Ws2  = (const float*)d_in[31];
  const float* bs2  = (const float*)d_in[32];
  float* out = (float*)d_out;

  // ---- workspace carve-up (fp32), with aliasing for the MoE phase ----
  float* f = (float*)d_ws;
  float* h     = f; f += ND;
  float* q     = f; f += ND;
  float* k     = f; f += ND;
  float* v     = f; f += ND;
  float* attnh = f; f += ND;
  float* x2    = f; f += ND;
  float* h2    = f; f += ND;
  float* acc   = f; f += ND;
  float* qc    = f; f += NQC;
  float* kc    = f; f += NQC;
  float* vc    = f; f += NQC;
  float* hs    = f; f += (size_t)NTOK * HIDd;
  float* gates = f; f += NTOK * NE;
  float* ew    = f; f += NE * NTOK;
  int*   cnt   = (int*)f;
  int*   eidx  = cnt + 64;
  // aliases (q/k/hs dead by MoE phase)
  float* h2g = q;
  float* heg = hs;
  float* oeg = k;

  const dim3 blk256(256), blk128(128);
  const dim3 gemmD(Dm / 128, NTOK / 128);     // M=1024 GEMMs
  const dim3 gemmH(HIDd / 128, NTOK / 128);   // M=4096 GEMMs

  // 0) init accumulators / counters
  zero_kernel<<<ND / 256, blk256, 0, stream>>>(acc, cnt);
  // 1) LN1
  ln_kernel<<<NTOK, blk256, 0, stream>>>(x, ln1g, ln1b, h);
  // 2) Q/K/V projections (WMMA, no bounds checks: rows == 2048)
  gemm_wmma<EP_STORE, false><<<gemmD, blk256, 0, stream>>>(h, Wq, bq, nullptr, q, NTOK, nullptr, Dm, Dm, 1.f);
  gemm_wmma<EP_STORE, false><<<gemmD, blk256, 0, stream>>>(h, Wk, bk, nullptr, k, NTOK, nullptr, Dm, Dm, 1.f);
  gemm_wmma<EP_STORE, false><<<gemmD, blk256, 0, stream>>>(h, Wv, bv, nullptr, v, NTOK, nullptr, Dm, Dm, 1.f);
  // 3) RoPE on q, k
  rope_kernel<<<(NTOK * Hn * 32) / 256, blk256, 0, stream>>>(q, cosb, sinb);
  rope_kernel<<<(NTOK * Hn * 32) / 256, blk256, 0, stream>>>(k, cosb, sinb);
  // 4) latent compressions
  latproj_kernel<<<NQC / 256, blk256, 0, stream>>>(q, Wqc, bqc, qc);
  latproj_kernel<<<NQC / 256, blk256, 0, stream>>>(k, Wkc, bkc, kc);
  latproj_kernel<<<NQC / 256, blk256, 0, stream>>>(v, Wvc, bvc, vc);
  // 5) causal latent attention (+ Wdc epilogue) -> attnh [NTOK, Dm]
  attn_kernel<<<dim3(Lctx / 128, Bq * Hn), blk128, 0, stream>>>(qc, kc, vc, Wdc, bdc, attnh);
  // 6) output projection + residual: x2 = x + attnh@Wo + bo
  gemm_wmma<EP_RESID, false><<<gemmD, blk256, 0, stream>>>(attnh, Wo, bo, x, x2, NTOK, nullptr, Dm, Dm, 1.f);
  // 7) LN2
  ln_kernel<<<NTOK, blk256, 0, stream>>>(x2, ln2g, ln2b, h2);
  // 8) shared experts: acc += mean_s( gelu(h2@Ws1_s + bs1) @ Ws2_s + bs2 )
  for (int s = 0; s < NS; ++s) {
    gemm_wmma<EP_GELU, false><<<gemmH, blk256, 0, stream>>>(
        h2, Ws1 + (size_t)s * Dm * HIDd, bs1 + s * HIDd, nullptr, hs, NTOK, nullptr, Dm, HIDd, 1.f);
    gemm_wmma<EP_ACCUM, false><<<gemmD, blk256, 0, stream>>>(
        hs, Ws2 + (size_t)s * HIDd * Dm, bs2 + s * Dm, nullptr, acc, NTOK, nullptr, HIDd, Dm, 1.f / NS);
  }
  // 9) router: gates -> top-2 -> per-expert compaction
  gate_kernel<<<(NTOK * NE) / 256, blk256, 0, stream>>>(h2, Wr, br, gates);
  topk_kernel<<<NTOK / 256, blk256, 0, stream>>>(gates, cnt, eidx, ew);
  // 10) routed experts (worst-case grids, device-side early exit on count)
  for (int e = 0; e < NE; ++e) {
    const int* ce = cnt + e;
    const int* ie = eidx + e * NTOK;
    const float* we = ew + e * NTOK;
    gather_kernel<<<dim3(Dm / 256, NTOK), blk256, 0, stream>>>(h2, ce, ie, h2g);
    gemm_wmma<EP_GELU, true><<<gemmH, blk256, 0, stream>>>(
        h2g, We1 + (size_t)e * Dm * HIDd, be1 + e * HIDd, nullptr, heg, NTOK, ce, Dm, HIDd, 1.f);
    gemm_wmma<EP_STORE, true><<<gemmD, blk256, 0, stream>>>(
        heg, We2 + (size_t)e * HIDd * Dm, be2 + e * Dm, nullptr, oeg, NTOK, ce, HIDd, Dm, 1.f);
    scatter_kernel<<<dim3(Dm / 256, NTOK), blk256, 0, stream>>>(oeg, ce, ie, we, acc);
  }
  // 11) final residual: out = x2 + (shared_mean + moe)
  final_add_kernel<<<ND / 256, blk256, 0, stream>>>(x2, acc, out);
}
